// TFBlipTextSelfAttention_23484881175189
// MI455X (gfx1250) — matrix operand order repaired and checked by reading, not compiled, not run on hardware
//
#include <hip/hip_runtime.h>
#include <cstdint>
#include <cstddef>

typedef __bf16 bf16_t;
typedef __attribute__((ext_vector_type(16))) __bf16 v16bf;
typedef __attribute__((ext_vector_type(8)))  float  v8f;

#define WMMA_BF16(a, b, c) \
  __builtin_amdgcn_wmma_f32_16x16x32_bf16(false, (a), false, (b), (short)0, (c), false, false)

union FragU { v16bf v; uint4 q[2]; };

__device__ __forceinline__ v8f v8f_zero() {
  v8f z;
#pragma unroll
  for (int i = 0; i < 8; ++i) z[i] = 0.0f;
  return z;
}

// ---- CDNA5 async copy: global -> LDS, 16B per lane, tracked by ASYNCcnt ---
__device__ __forceinline__ void async_ld_b128(void* lds, const void* gptr) {
  uint32_t off = (uint32_t)(uintptr_t)lds;  // low 32 bits of flat addr = LDS offset
  unsigned long long ga = (unsigned long long)(uintptr_t)gptr;
  asm volatile("global_load_async_to_lds_b128 %0, %1, off"
               :: "v"(off), "v"(ga) : "memory");
}
__device__ __forceinline__ void wait_async() {
  asm volatile("s_wait_asynccnt 0x0" ::: "memory");
}

// A-matrix fragment (16x32 bf16 slice of a row-major tile, stride in elems).
// Lane L: row M = L%16, K-base kb = (L/16)*8; elems 0..7 <- K kb..kb+7,
// elems 8..15 <- K kb+16..kb+23  (two 16B chunks).
__device__ __forceinline__ v16bf load_frag_a(const bf16_t* base, int stride) {
  const int lane = threadIdx.x & 31;
  const int row  = lane & 15;
  const int kb   = (lane >> 4) << 3;
  const bf16_t* p = base + row * stride + kb;
  FragU f;
  f.q[0] = *(const uint4*)(p);
  f.q[1] = *(const uint4*)(p + 16);
  return f.v;
}

// B-matrix fragment, consumed as B^T stored row-major [N][K], K=32 slice.
// Lane L: N = L%16, K-base kh = (L/16)*8; elems 0..7 <- K kh..kh+7,
// elems 8..15 <- K kh+16..kh+23  (two 16B chunks).
__device__ __forceinline__ v16bf load_frag_b(const bf16_t* base, int stride) {
  const int lane = threadIdx.x & 31;
  const int n    = lane & 15;
  const int kh   = (lane >> 4) << 3;
  const bf16_t* p = base + n * stride + kh;
  FragU f;
  f.q[0] = *(const uint4*)(p);
  f.q[1] = *(const uint4*)(p + 16);
  return f.v;
}

// ---------------- Kernel 1: fp32 -> bf16 convert ----------------
__global__ void cvt_f32_to_bf16_kernel(const float* __restrict__ in,
                                       bf16_t* __restrict__ out, int n) {
  int i = (blockIdx.x * blockDim.x + threadIdx.x) * 4;
  if (i + 3 < n) {
    float4 v = *(const float4*)(in + i);
    out[i + 0] = (bf16_t)v.x;
    out[i + 1] = (bf16_t)v.y;
    out[i + 2] = (bf16_t)v.z;
    out[i + 3] = (bf16_t)v.w;
  }
}

// ---------------- Kernel 2: pack W (1024x1024 f32) -> WT (bf16, [n][k]) ---
__global__ void transpose_w_kernel(const float* __restrict__ W,
                                   bf16_t* __restrict__ WTo) {
  int idx = blockIdx.x * blockDim.x + threadIdx.x;  // 1M outputs
  int n = idx >> 10, k = idx & 1023;
  WTo[idx] = (bf16_t)W[k * 1024 + n];
}

// ---------------- Kernel 3: fused QKV GEMM --------------------------------
// Block tile 128(M) x 64(N), 4 waves, each wave owns 32x64.
// K staged 64-deep, double-buffered LDS fed by async global->LDS copies:
// 16 WMMA per barrier, fragment loads batched ahead of each WMMA burst.
__global__ __launch_bounds__(128)
void qkv_gemm_kernel(const bf16_t* __restrict__ X,   // [8192,1024]
                     const bf16_t* __restrict__ WT,  // [3072,1024] ([n][k])
                     const float* __restrict__ bq,
                     const float* __restrict__ bk,
                     const float* __restrict__ bv,
                     bf16_t* __restrict__ Qb,   // [64][2048][64], pre-scaled
                     bf16_t* __restrict__ Kb,   // [64][2048][64]
                     bf16_t* __restrict__ Vt) { // [64][64][2048]
  __shared__ __align__(16) bf16_t Atile[2][128 * 64];  // 2 x 16KB
  __shared__ __align__(16) bf16_t Btile[2][64 * 64];   // 2 x  8KB

  const int tid  = threadIdx.x;
  const int wave = tid >> 5;
  const int lane = tid & 31;
  const int m0 = blockIdx.x * 128;
  const int n0 = blockIdx.y * 64;

  v8f acc0[4], acc1[4];
#pragma unroll
  for (int i = 0; i < 4; ++i) { acc0[i] = v8f_zero(); acc1[i] = v8f_zero(); }

  // stage buffer `sel` with a 64-deep k-slice at k0
  auto stage = [&](int sel, int k0) {
#pragma unroll
    for (int i = 0; i < 8; ++i) {  // A: 128 rows x 8 chunks = 1024
      int c = tid + i * 128;
      int row = c >> 3, seg = c & 7;
      async_ld_b128(&Atile[sel][row * 64 + seg * 8],
                    X + (size_t)(m0 + row) * 1024 + k0 + seg * 8);
    }
#pragma unroll
    for (int i = 0; i < 4; ++i) {  // B: 64 rows x 8 chunks = 512
      int c = tid + i * 128;
      int row = c >> 3, seg = c & 7;
      async_ld_b128(&Btile[sel][row * 64 + seg * 8],
                    WT + (size_t)(n0 + row) * 1024 + k0 + seg * 8);
    }
  };

  stage(0, 0);
  wait_async();
  __syncthreads();

  for (int k0 = 0; k0 < 1024; k0 += 64) {
    const int cur = (k0 >> 6) & 1;
    if (k0 + 64 < 1024) stage(cur ^ 1, k0 + 64);

    const bf16_t* As = &Atile[cur][0];
    const bf16_t* Bs = &Btile[cur][0];
#pragma unroll
    for (int kk = 0; kk < 64; kk += 32) {
      // batch all 6 fragment loads, then an 8-WMMA burst
      v16bf a0 = load_frag_a(As + (wave * 32 + 0) * 64 + kk, 64);
      v16bf a1 = load_frag_a(As + (wave * 32 + 16) * 64 + kk, 64);
      v16bf b0 = load_frag_b(Bs + 0 * 16 * 64 + kk, 64);
      v16bf b1 = load_frag_b(Bs + 1 * 16 * 64 + kk, 64);
      v16bf b2 = load_frag_b(Bs + 2 * 16 * 64 + kk, 64);
      v16bf b3 = load_frag_b(Bs + 3 * 16 * 64 + kk, 64);
      acc0[0] = WMMA_BF16(a0, b0, acc0[0]);
      acc1[0] = WMMA_BF16(a1, b0, acc1[0]);
      acc0[1] = WMMA_BF16(a0, b1, acc0[1]);
      acc1[1] = WMMA_BF16(a1, b1, acc1[1]);
      acc0[2] = WMMA_BF16(a0, b2, acc0[2]);
      acc1[2] = WMMA_BF16(a1, b2, acc1[2]);
      acc0[3] = WMMA_BF16(a0, b3, acc0[3]);
      acc1[3] = WMMA_BF16(a1, b3, acc1[3]);
    }
    wait_async();
    __syncthreads();
  }

  // epilogue: bias add + scatter in bf16 (Q pre-scaled by 1/sqrt(dh))
  const int colN = lane & 15;
  const int rsel = (lane >> 4) * 8;
#pragma unroll
  for (int nt = 0; nt < 4; ++nt) {
    int ng    = n0 + nt * 16 + colN;  // 0..3071
    int which = ng >> 10;
    int nidx  = ng & 1023;
    int h = nidx >> 6, dd = nidx & 63;
    float bias = (which == 0 ? bq : which == 1 ? bk : bv)[nidx];
#pragma unroll
    for (int ms = 0; ms < 2; ++ms) {
      const v8f& a = ms ? acc1[nt] : acc0[nt];
#pragma unroll
      for (int r = 0; r < 8; ++r) {
        int m = m0 + wave * 32 + ms * 16 + r + rsel;  // 0..8191
        int b = m >> 11, s = m & 2047;
        int bh = b * 16 + h;
        float v = a[r] + bias;
        if (which == 0)      Qb[((size_t)bh * 2048 + s) * 64 + dd] = (bf16_t)(v * 0.125f);
        else if (which == 1) Kb[((size_t)bh * 2048 + s) * 64 + dd] = (bf16_t)v;
        else                 Vt[((size_t)bh * 64 + dd) * 2048 + s] = (bf16_t)v;
      }
    }
  }
}

// ---------------- Kernel 4: flash attention -------------------------------
// kv staged 64-wide (4 score tiles), double-buffered async K/V:
// 16 WMMA per barrier, fragment loads batched ahead of the WMMA bursts.
__global__ __launch_bounds__(128)
void flash_attn_kernel(const bf16_t* __restrict__ Qb,
                       const bf16_t* __restrict__ Kb,
                       const bf16_t* __restrict__ Vt,
                       const float* __restrict__ mask,  // [B,2048]
                       float* __restrict__ out) {       // [B,2048,1024]
  __shared__ __align__(16) bf16_t Ktile[2][64 * 64];  // 2 x 8KB  [kv][dh]
  __shared__ __align__(16) bf16_t Vtile[2][64 * 64];  // 2 x 8KB  [dh][kv]
  __shared__ __align__(16) bf16_t Pbuf[4][16 * 64];   // 8KB, wave-private

  const int tid  = threadIdx.x;
  const int wave = tid >> 5;
  const int lane = tid & 31;
  const int qblk = blockIdx.x;  // 32 blocks of 64 q rows
  const int bh   = blockIdx.y;  // 64
  const int b    = bh >> 4;
  const int h    = bh & 15;

  const bf16_t* Kbh = Kb + (size_t)bh * 2048 * 64;
  const bf16_t* Vbh = Vt + (size_t)bh * 64 * 2048;

  const int q0 = qblk * 64 + wave * 16;
  const bf16_t* Qbase = Qb + ((size_t)bh * 2048 + q0) * 64;

  // Q fragments for dh slices [0,32) and [32,64) (Q is pre-scaled)
  v16bf qf0, qf1;
  {
    const int row = lane & 15;
    const int kb  = (lane >> 4) << 3;
    const bf16_t* p = Qbase + row * 64;
    FragU f0, f1;
    f0.q[0] = *(const uint4*)(p + kb);
    f0.q[1] = *(const uint4*)(p + kb + 16);
    f1.q[0] = *(const uint4*)(p + 32 + kb);
    f1.q[1] = *(const uint4*)(p + 32 + kb + 16);
    qf0 = f0.v;
    qf1 = f1.v;
  }

  float m_i[8], l_i[8];
  v8f acc[4];
#pragma unroll
  for (int r = 0; r < 8; ++r) { m_i[r] = -3.0e38f; l_i[r] = 0.0f; }
#pragma unroll
  for (int i = 0; i < 4; ++i) acc[i] = v8f_zero();

  const int colN = lane & 15;
  const int rsel = (lane >> 4) * 8;

  auto stage = [&](int sel, int kv0) {
#pragma unroll
    for (int i = 0; i < 4; ++i) {  // K: 64 rows x 8 chunks = 512
      int c = tid + i * 128;
      int row = c >> 3, seg = c & 7;
      async_ld_b128(&Ktile[sel][row * 64 + seg * 8],
                    Kbh + (size_t)(kv0 + row) * 64 + seg * 8);
    }
#pragma unroll
    for (int i = 0; i < 4; ++i) {  // V^T: 64 dh-rows x 8 kv-chunks = 512
      int c = tid + i * 128;
      int row = c >> 3, seg = c & 7;
      async_ld_b128(&Vtile[sel][row * 64 + seg * 8],
                    Vbh + (size_t)row * 2048 + kv0 + seg * 8);
    }
  };

  stage(0, 0);
  wait_async();
  __syncthreads();

  for (int kv0 = 0; kv0 < 2048; kv0 += 64) {
    const int cur = (kv0 >> 6) & 1;
    if (kv0 + 64 < 2048) stage(cur ^ 1, kv0 + 64);

    const bf16_t* Ks = &Ktile[cur][0];
    const bf16_t* Vs = &Vtile[cur][0];

    // scores: four 16x16 tiles over this kv block, K-dim = dh = 64 (2 steps);
    // per-tile fragment pair batched before its 2-WMMA burst
    v8f s[4];
#pragma unroll
    for (int j = 0; j < 4; ++j) {
      v16bf kf0 = load_frag_b(Ks + j * 16 * 64 + 0, 64);
      v16bf kf1 = load_frag_b(Ks + j * 16 * 64 + 32, 64);
      s[j] = v8f_zero();
      s[j] = WMMA_BF16(qf0, kf0, s[j]);
      s[j] = WMMA_BF16(qf1, kf1, s[j]);
    }

#pragma unroll
    for (int j = 0; j < 4; ++j) {
      float mv = mask[b * 2048 + kv0 + j * 16 + colN];
#pragma unroll
      for (int r = 0; r < 8; ++r) s[j][r] += mv;  // Q pre-scaled: just mask
    }

    // online softmax; C-layout rows at VGPR r (+8 for upper lane half),
    // 16 columns spread across a 16-lane half -> xor-reduce with masks 1..8
    bf16_t* pb = &Pbuf[wave][0];
#pragma unroll
    for (int r = 0; r < 8; ++r) {
      float mx = fmaxf(fmaxf(s[0][r], s[1][r]), fmaxf(s[2][r], s[3][r]));
      mx = fmaxf(mx, __shfl_xor(mx, 1));
      mx = fmaxf(mx, __shfl_xor(mx, 2));
      mx = fmaxf(mx, __shfl_xor(mx, 4));
      mx = fmaxf(mx, __shfl_xor(mx, 8));
      float mnew = fmaxf(m_i[r], mx);
      float corr = __expf(m_i[r] - mnew);
      float p0 = __expf(s[0][r] - mnew);
      float p1 = __expf(s[1][r] - mnew);
      float p2 = __expf(s[2][r] - mnew);
      float p3 = __expf(s[3][r] - mnew);
      float rs = (p0 + p1) + (p2 + p3);
      rs += __shfl_xor(rs, 1);
      rs += __shfl_xor(rs, 2);
      rs += __shfl_xor(rs, 4);
      rs += __shfl_xor(rs, 8);
      l_i[r] = l_i[r] * corr + rs;
      m_i[r] = mnew;
#pragma unroll
      for (int nt = 0; nt < 4; ++nt) acc[nt][r] *= corr;
      // spill P (C-layout -> row-major 16x64 bf16) for the PV A-fragments
      int ro = (r + rsel) * 64 + colN;
      pb[ro + 0]  = (bf16_t)p0;
      pb[ro + 16] = (bf16_t)p1;
      pb[ro + 32] = (bf16_t)p2;
      pb[ro + 48] = (bf16_t)p3;
    }

    v16bf pf0 = load_frag_a(pb + 0, 64);   // P, kv slice [0,32)
    v16bf pf1 = load_frag_a(pb + 32, 64);  // P, kv slice [32,64)
#pragma unroll
    for (int nt = 0; nt < 4; ++nt) {
      v16bf vf0 = load_frag_b(Vs + nt * 16 * 64 + 0, 64);
      v16bf vf1 = load_frag_b(Vs + nt * 16 * 64 + 32, 64);
      acc[nt] = WMMA_BF16(pf0, vf0, acc[nt]);
      acc[nt] = WMMA_BF16(pf1, vf1, acc[nt]);
    }
    wait_async();
    __syncthreads();
  }

  // finalize: divide by l, write fp32 out[B,S,H*dh]
#pragma unroll
  for (int r = 0; r < 8; ++r) {
    float inv = 1.0f / l_i[r];
    int srow = q0 + r + rsel;
    float* orow = out + ((size_t)b * 2048 + srow) * 1024 + h * 64;
#pragma unroll
    for (int nt = 0; nt < 4; ++nt)
      orow[nt * 16 + colN] = acc[nt][r] * inv;
  }
}

// ---------------- launch ---------------------------------------------------
extern "C" void kernel_launch(void* const* d_in, const int* in_sizes, int n_in,
                              void* d_out, int out_size, void* d_ws, size_t ws_size,
                              hipStream_t stream) {
  const float* hidden = (const float*)d_in[0];
  const float* mask   = (const float*)d_in[1];
  const float* Wq     = (const float*)d_in[2];
  const float* bq     = (const float*)d_in[3];
  const float* Wk     = (const float*)d_in[4];
  const float* bk     = (const float*)d_in[5];
  const float* Wv     = (const float*)d_in[6];
  const float* bv     = (const float*)d_in[7];
  float* out = (float*)d_out;

  char* ws = (char*)d_ws;
  size_t off = 0;
  bf16_t* Xb = (bf16_t*)(ws + off); off += (size_t)8192 * 1024 * 2;  // 16 MB
  bf16_t* WT = (bf16_t*)(ws + off); off += (size_t)3072 * 1024 * 2;  //  6 MB
  bf16_t* Qb = (bf16_t*)(ws + off); off += (size_t)8192 * 1024 * 2;  // 16 MB
  bf16_t* Kb = (bf16_t*)(ws + off); off += (size_t)8192 * 1024 * 2;  // 16 MB
  bf16_t* Vt = (bf16_t*)(ws + off);                                  // 16 MB

  cvt_f32_to_bf16_kernel<<<8192, 256, 0, stream>>>(hidden, Xb, 8192 * 1024);
  transpose_w_kernel<<<4096, 256, 0, stream>>>(Wq, WT + (size_t)0 * 1024 * 1024);
  transpose_w_kernel<<<4096, 256, 0, stream>>>(Wk, WT + (size_t)1 * 1024 * 1024);
  transpose_w_kernel<<<4096, 256, 0, stream>>>(Wv, WT + (size_t)2 * 1024 * 1024);

  qkv_gemm_kernel<<<dim3(64, 48), 128, 0, stream>>>(Xb, WT, bq, bk, bv, Qb, Kb, Vt);
  flash_attn_kernel<<<dim3(32, 64), 128, 0, stream>>>(Qb, Kb, Vt, mask, out);
}
